// ComENetModel_33818572488722
// MI455X (gfx1250) — compile-verified, matrix operand order checked by reading, not run on hardware
//
#include <hip/hip_runtime.h>

// ---------------------------------------------------------------------------
// ComENet-style GNN forward for MI455X (gfx1250, wave32, WMMA).
// GEMMs run on v_wmma_f32_16x16x32_f16 (f16 in, f32 accumulate).
// Weights are pre-swizzled into WMMA B-fragment order, staged once per block
// into LDS, and held in registers across a grid-stride tile loop.
// ---------------------------------------------------------------------------

typedef _Float16 v16h __attribute__((ext_vector_type(16)));
typedef _Float16 v8h  __attribute__((ext_vector_type(8)));
typedef float    v8f  __attribute__((ext_vector_type(8)));

#define EPSF 1e-8f

__device__ __forceinline__ float clampf(float x, float lo, float hi) {
    return fminf(fmaxf(x, lo), hi);
}

// ---------------------------------------------------------------------------
// WMMA fragment layouts (wave32, 16x16x32 f16 shapes).
// A: 16x32, row M = lane%16; half = lane/16 selects K sub-bands:
//    elems 0..7  -> K = 8*half + i        (+k0)
//    elems 8..15 -> K = 16 + 8*half + i-8 (+k0)
// B: 32x16, col N = lane%16; same K banding (mirror of A).
// C/D: 16x16 f32, lane holds col N = lane%16, VGPR r holds row M = r + 8*half.
// ---------------------------------------------------------------------------
__device__ __forceinline__ v16h load_a_frag(const _Float16* __restrict__ M,
                                            int row0, int k0, int lane) {
    int m = lane & 15, half = lane >> 4;
    const _Float16* p = M + (size_t)(row0 + m) * 64 + k0 + 8 * half;
    v8h c0 = *(const v8h*)(p);
    v8h c1 = *(const v8h*)(p + 16);
    v16h a;
#pragma unroll
    for (int i = 0; i < 8; ++i) { a[i] = c0[i]; a[8 + i] = c1[i]; }
    return a;
}

// ---------------------------------------------------------------------------
// Repack a row-major f32 weight W[64][64] (K-major) into f16 B-fragment
// order: Wp[((nt*2+kt)*32 + lane)*16 + i], so each lane's fragment is two
// contiguous 16B loads.
// ---------------------------------------------------------------------------
__global__ void k_pack_weight(const float* __restrict__ W,
                              _Float16* __restrict__ Wp) {
    int o = blockIdx.x * blockDim.x + threadIdx.x;
    if (o >= 4096) return;
    int i = o & 15;
    int l = (o >> 4) & 31;
    int f = o >> 9;               // f = nt*2 + kt
    int kt = f & 1, nt = f >> 1;
    int half = l >> 4, n = nt * 16 + (l & 15);
    int k = kt * 32 + ((i < 8) ? (8 * half + i) : (16 + 8 * half + (i - 8)));
    Wp[o] = (_Float16)W[k * 64 + n];
}

// ---------------------------------------------------------------------------
// Node GEMM: out[n,64] = A[n,64](f16) @ W + bias, RELU compile-time.
// Packed weights staged in LDS once per block; 8 B fragments pinned in
// registers; grid-stride loop over 16-row tiles (8 waves / block).
// ---------------------------------------------------------------------------
template <bool RELU>
__global__ __launch_bounds__(256) void node_gemm_f16(
    const _Float16* __restrict__ A, const _Float16* __restrict__ Wp,
    const float* __restrict__ bias, float* __restrict__ out, int n)
{
    __shared__ __align__(16) _Float16 sW[4096];
    __shared__ float sB[64];
    int t = threadIdx.x;
    ((v8h*)sW)[t]       = ((const v8h*)Wp)[t];
    ((v8h*)sW)[t + 256] = ((const v8h*)Wp)[t + 256];
    if (t < 64) sB[t] = bias[t];
    __syncthreads();

    int lane = t & 31, wave = t >> 5;
    int ncol = lane & 15, half = lane >> 4;

    v16h B[8];
#pragma unroll
    for (int f = 0; f < 8; ++f)
        B[f] = *(const v16h*)(sW + (f * 32 + lane) * 16);
    float bn[4];
#pragma unroll
    for (int nt = 0; nt < 4; ++nt) bn[nt] = sB[nt * 16 + ncol];

    int tiles = (n + 15) >> 4;
    int nwaves = gridDim.x * 8;
    for (int tile = blockIdx.x * 8 + wave; tile < tiles; tile += nwaves) {
        int row0 = tile * 16;
        int next = tile + nwaves;
        if (next < tiles)
            __builtin_prefetch(A + (size_t)next * 16 * 64, 0, 1);
        v16h a0 = load_a_frag(A, row0, 0, lane);
        v16h a1 = load_a_frag(A, row0, 32, lane);
#pragma unroll
        for (int nt = 0; nt < 4; ++nt) {
            v8f c;
#pragma unroll
            for (int r = 0; r < 8; ++r) c[r] = bn[nt];
            c = __builtin_amdgcn_wmma_f32_16x16x32_f16(false, a0, false,
                    B[nt * 2 + 0], (short)0, c, false, false);
            c = __builtin_amdgcn_wmma_f32_16x16x32_f16(false, a1, false,
                    B[nt * 2 + 1], (short)0, c, false, false);
#pragma unroll
            for (int r = 0; r < 8; ++r) {
                float v = c[r];
                if (RELU) v = fmaxf(v, 0.f);
                out[(size_t)(row0 + r + 8 * half) * 64 + nt * 16 + ncol] = v;
            }
        }
    }
}

// ---------------------------------------------------------------------------
// Fused edge conv: per 16-edge tile,
//   hidden = relu(geo @ aw + ab)   (built in-register from LDS-staged aw/ab)
//   ge     = hidden @ bw + bb      (WMMA, f32 accum, B frags pinned in regs)
//   agg[dst] += ge + hlin[src]     (fused gather + atomic scatter)
// ---------------------------------------------------------------------------
__global__ __launch_bounds__(256) void edge_conv_wmma(
    const float* __restrict__ geo, const float* __restrict__ aw,
    const float* __restrict__ ab, const _Float16* __restrict__ bwp,
    const float* __restrict__ bb, const float* __restrict__ hlin,
    const int* __restrict__ src, const int* __restrict__ dst,
    float* __restrict__ agg, int e)
{
    __shared__ __align__(16) _Float16 sW[4096];
    __shared__ float sAW[256];
    __shared__ float sAB[64];
    __shared__ float sBB[64];
    int t = threadIdx.x;
    ((v8h*)sW)[t]       = ((const v8h*)bwp)[t];
    ((v8h*)sW)[t + 256] = ((const v8h*)bwp)[t + 256];
    sAW[t] = aw[t];
    if (t < 64) { sAB[t] = ab[t]; sBB[t] = bb[t]; }
    __syncthreads();

    int lane = t & 31, wave = t >> 5;
    int m = lane & 15, half = lane >> 4;
    int ncol = lane & 15;

    v16h B[8];
#pragma unroll
    for (int f = 0; f < 8; ++f)
        B[f] = *(const v16h*)(sW + (f * 32 + lane) * 16);
    float bn[4];
#pragma unroll
    for (int nt = 0; nt < 4; ++nt) bn[nt] = sBB[nt * 16 + ncol];

    int tiles = (e + 15) >> 4;
    int nwaves = gridDim.x * 8;
    for (int tile = blockIdx.x * 8 + wave; tile < tiles; tile += nwaves) {
        int e0 = tile * 16;
        int next = tile + nwaves;
        if (next < tiles)
            __builtin_prefetch(geo + (size_t)next * 16 * 4, 0, 1);

        const float* g = geo + (size_t)(e0 + m) * 4;
        float g0 = g[0], g1 = g[1], g2 = g[2], g3 = g[3];

        v16h a0, a1;
#pragma unroll
        for (int i = 0; i < 16; ++i) {
            int kk = (i < 8) ? (8 * half + i) : (16 + 8 * half + (i - 8));
            float h0 = fmaxf(g0 * sAW[kk] + g1 * sAW[64 + kk] +
                             g2 * sAW[128 + kk] + g3 * sAW[192 + kk] +
                             sAB[kk], 0.f);
            int k1 = kk + 32;
            float h1 = fmaxf(g0 * sAW[k1] + g1 * sAW[64 + k1] +
                             g2 * sAW[128 + k1] + g3 * sAW[192 + k1] +
                             sAB[k1], 0.f);
            a0[i] = (_Float16)h0;
            a1[i] = (_Float16)h1;
        }

        int se[8], de[8];
#pragma unroll
        for (int r = 0; r < 8; ++r) {
            int ed = e0 + r + 8 * half;
            se[r] = src[ed];
            de[r] = dst[ed];
        }

#pragma unroll
        for (int nt = 0; nt < 4; ++nt) {
            v8f c;
#pragma unroll
            for (int r = 0; r < 8; ++r) c[r] = bn[nt];
            c = __builtin_amdgcn_wmma_f32_16x16x32_f16(false, a0, false,
                    B[nt * 2 + 0], (short)0, c, false, false);
            c = __builtin_amdgcn_wmma_f32_16x16x32_f16(false, a1, false,
                    B[nt * 2 + 1], (short)0, c, false, false);
#pragma unroll
            for (int r = 0; r < 8; ++r) {
                float msg = c[r] + hlin[(size_t)se[r] * 64 + nt * 16 + ncol];
                atomicAdd(&agg[(size_t)de[r] * 64 + nt * 16 + ncol], msg);
            }
        }
    }
}

// ---------------------------------------------------------------------------
// Small utility kernels
// ---------------------------------------------------------------------------
__global__ void k_fill_u32(unsigned* __restrict__ p, unsigned v, int n) {
    int i = blockIdx.x * blockDim.x + threadIdx.x;
    if (i < n) p[i] = v;
}

__global__ void k_f32_to_f16(const float* __restrict__ s,
                             _Float16* __restrict__ d, int n) {
    int i = blockIdx.x * blockDim.x + threadIdx.x;
    if (i < n) d[i] = (_Float16)s[i];
}

__global__ void k_relu_f32_to_f16(const float* __restrict__ s,
                                  _Float16* __restrict__ d, int n) {
    int i = blockIdx.x * blockDim.x + threadIdx.x;
    if (i < n) d[i] = (_Float16)fmaxf(s[i], 0.f);
}

// ---------------------------------------------------------------------------
// Nearest / 2nd-nearest out-neighbor search (float bits are order-preserving
// for non-negative values, so uint atomicMin == float segment_min).
// ---------------------------------------------------------------------------
__global__ void k_edge_dist(const float* __restrict__ pos,
                            const int* __restrict__ src,
                            const int* __restrict__ dst,
                            unsigned* __restrict__ dbits,
                            unsigned* __restrict__ m1,
                            int* __restrict__ deg, int e)
{
    int i = blockIdx.x * blockDim.x + threadIdx.x;
    if (i >= e) return;
    int s = src[i], t = dst[i];
    float dx = pos[3 * s + 0] - pos[3 * t + 0];
    float dy = pos[3 * s + 1] - pos[3 * t + 1];
    float dz = pos[3 * s + 2] - pos[3 * t + 2];
    float d = sqrtf(dx * dx + dy * dy + dz * dz);
    unsigned b = __float_as_uint(d);
    dbits[i] = b;
    atomicMin(&m1[s], b);
    atomicAdd(&deg[s], 1);
}

__global__ void k_e1(const unsigned* __restrict__ dbits,
                     const int* __restrict__ src,
                     const unsigned* __restrict__ m1,
                     unsigned* __restrict__ e1, int e)
{
    int i = blockIdx.x * blockDim.x + threadIdx.x;
    if (i >= e) return;
    int s = src[i];
    if (dbits[i] <= m1[s]) atomicMin(&e1[s], (unsigned)i);
}

__global__ void k_m2(const unsigned* __restrict__ dbits,
                     const int* __restrict__ src,
                     const unsigned* __restrict__ e1,
                     unsigned* __restrict__ m2, int e)
{
    int i = blockIdx.x * blockDim.x + threadIdx.x;
    if (i >= e) return;
    int s = src[i];
    if ((unsigned)i != e1[s]) atomicMin(&m2[s], dbits[i]);
}

__global__ void k_e2(const unsigned* __restrict__ dbits,
                     const int* __restrict__ src,
                     const unsigned* __restrict__ e1,
                     const unsigned* __restrict__ m2,
                     unsigned* __restrict__ e2, int e)
{
    int i = blockIdx.x * blockDim.x + threadIdx.x;
    if (i >= e) return;
    int s = src[i];
    if ((unsigned)i != e1[s] && dbits[i] <= m2[s]) atomicMin(&e2[s], (unsigned)i);
}

__global__ void k_nn(const unsigned* __restrict__ e1,
                     const unsigned* __restrict__ e2,
                     const int* __restrict__ dst,
                     int* __restrict__ nnA, int* __restrict__ nnB,
                     int n, int e)
{
    int i = blockIdx.x * blockDim.x + threadIdx.x;
    if (i >= n) return;
    unsigned a = e1[i], b = e2[i];
    nnA[i] = (a < (unsigned)e) ? dst[a] : i;
    nnB[i] = (b < (unsigned)e) ? dst[b] : i;
}

// ---------------------------------------------------------------------------
// Per-edge geometry: d, theta, phi, tau
// ---------------------------------------------------------------------------
__global__ void k_geo(const float* __restrict__ pos,
                      const int* __restrict__ src, const int* __restrict__ dst,
                      const int* __restrict__ nnA, const int* __restrict__ nnB,
                      const int* __restrict__ deg,
                      float* __restrict__ geo, int e)
{
    int i = blockIdx.x * blockDim.x + threadIdx.x;
    if (i >= e) return;
    int s = src[i], t = dst[i];
    int nAs = nnA[s];
    int fi = (nAs != t) ? nAs : ((deg[s] >= 2) ? nnB[s] : s);
    int nAt = nnA[t];
    int fj = (nAt != s) ? nAt : ((deg[t] >= 2) ? nnB[t] : t);

    float sx = pos[3 * s], sy = pos[3 * s + 1], sz = pos[3 * s + 2];
    float tx = pos[3 * t], ty = pos[3 * t + 1], tz = pos[3 * t + 2];
    float rx = tx - sx, ry = ty - sy, rz = tz - sz;
    float d = sqrtf(rx * rx + ry * ry + rz * rz);
    float theta = acosf(clampf(rz / (d + EPSF), -1.f + EPSF, 1.f - EPSF));
    float phi = atan2f(ry, rx);

    float v1x = sx - pos[3 * fi], v1y = sy - pos[3 * fi + 1], v1z = sz - pos[3 * fi + 2];
    float v3x = tx - pos[3 * fj], v3y = ty - pos[3 * fj + 1], v3z = tz - pos[3 * fj + 2];
    // n1 = v1 x rel ; n2 = rel x v3
    float n1x = v1y * rz - v1z * ry, n1y = v1z * rx - v1x * rz, n1z = v1x * ry - v1y * rx;
    float n2x = ry * v3z - rz * v3y, n2y = rz * v3x - rx * v3z, n2z = rx * v3y - ry * v3x;
    float l1n = sqrtf(n1x * n1x + n1y * n1y + n1z * n1z + 1e-12f) + EPSF;
    float l2n = sqrtf(n2x * n2x + n2y * n2y + n2z * n2z + 1e-12f) + EPSF;
    float ct = clampf((n1x * n2x + n1y * n2y + n1z * n2z) / (l1n * l2n),
                      -1.f + EPSF, 1.f - EPSF);
    float tau = acosf(ct);

    geo[(size_t)i * 4 + 0] = d;
    geo[(size_t)i * 4 + 1] = theta;
    geo[(size_t)i * 4 + 2] = phi;
    geo[(size_t)i * 4 + 3] = tau;
}

// ---------------------------------------------------------------------------
// Pooling + head
// ---------------------------------------------------------------------------
__global__ void k_pool_cnt(const int* __restrict__ batch, int* __restrict__ cnt,
                           int n) {
    int i = blockIdx.x * blockDim.x + threadIdx.x;
    if (i < n) atomicAdd(&cnt[batch[i]], 1);
}

__global__ void k_pool_sum(const float* __restrict__ hs,
                           const int* __restrict__ batch,
                           float* __restrict__ gsum, int n) {
    int idx = blockIdx.x * blockDim.x + threadIdx.x;
    if (idx >= n * 64) return;
    int i = idx >> 6, f = idx & 63;
    atomicAdd(&gsum[(size_t)batch[i] * 64 + f], hs[idx]);
}

__global__ void k_final(const float* __restrict__ gsum,
                        const int* __restrict__ cnt,
                        const float* __restrict__ l1w,
                        const float* __restrict__ l1b,
                        const float* __restrict__ l2w,
                        const float* __restrict__ l2b,
                        float* __restrict__ out, int G)
{
    int g = blockIdx.x * blockDim.x + threadIdx.x;
    if (g >= G) return;
    float inv = 1.f / fmaxf((float)cnt[g], 1.f);
    float mean[64];
#pragma unroll
    for (int k = 0; k < 64; ++k) mean[k] = gsum[(size_t)g * 64 + k] * inv;
    float acc = l2b[0];
    for (int j = 0; j < 32; ++j) {
        float z = l1b[j];
        for (int k = 0; k < 64; ++k) z += mean[k] * l1w[k * 32 + j];
        acc += fmaxf(z, 0.f) * l2w[j];
    }
    out[g] = acc;
}

// ---------------------------------------------------------------------------
// Host-side orchestration
// ---------------------------------------------------------------------------
extern "C" void kernel_launch(void* const* d_in, const int* in_sizes, int n_in,
                              void* d_out, int out_size, void* d_ws, size_t ws_size,
                              hipStream_t stream)
{
    const float* x    = (const float*)d_in[0];
    const int*   ei   = (const int*)d_in[1];
    const int*   batch= (const int*)d_in[2];
    const float* pos  = (const float*)d_in[3];
    const float* c1lw = (const float*)d_in[4];  const float* c1lb = (const float*)d_in[5];
    const float* c1aw = (const float*)d_in[6];  const float* c1ab = (const float*)d_in[7];
    const float* c1bw = (const float*)d_in[8];  const float* c1bb = (const float*)d_in[9];
    const float* c2lw = (const float*)d_in[10]; const float* c2lb = (const float*)d_in[11];
    const float* c2aw = (const float*)d_in[12]; const float* c2ab = (const float*)d_in[13];
    const float* c2bw = (const float*)d_in[14]; const float* c2bb = (const float*)d_in[15];
    const float* saw  = (const float*)d_in[16]; const float* sab  = (const float*)d_in[17];
    const float* l1w  = (const float*)d_in[18]; const float* l1b  = (const float*)d_in[19];
    const float* l2w  = (const float*)d_in[20]; const float* l2b  = (const float*)d_in[21];

    const int N = in_sizes[2];
    const int E = in_sizes[1] / 2;
    const int G = out_size;
    const int* src = ei;
    const int* dst = ei + E;

    // Workspace carve-up (256B-aligned)
    char* w = (char*)d_ws;
    auto alloc = [&](size_t bytes) -> char* {
        char* p = w;
        w += (bytes + 255) & ~(size_t)255;
        return p;
    };
    unsigned* dbits = (unsigned*)alloc((size_t)E * 4);
    unsigned* m1    = (unsigned*)alloc((size_t)N * 4);
    unsigned* e1    = (unsigned*)alloc((size_t)N * 4);
    unsigned* m2    = (unsigned*)alloc((size_t)N * 4);
    unsigned* e2    = (unsigned*)alloc((size_t)N * 4);
    int*      deg   = (int*)alloc((size_t)N * 4);
    int*      nnA   = (int*)alloc((size_t)N * 4);
    int*      nnB   = (int*)alloc((size_t)N * 4);
    float*    geo   = (float*)alloc((size_t)E * 4 * 4);
    _Float16* hh    = (_Float16*)alloc((size_t)N * 64 * 2);
    _Float16* wp_c1lw = (_Float16*)alloc(64 * 64 * 2);
    _Float16* wp_c1bw = (_Float16*)alloc(64 * 64 * 2);
    _Float16* wp_c2lw = (_Float16*)alloc(64 * 64 * 2);
    _Float16* wp_c2bw = (_Float16*)alloc(64 * 64 * 2);
    _Float16* wp_saw  = (_Float16*)alloc(64 * 64 * 2);
    float*    hlin  = (float*)alloc((size_t)N * 64 * 4);
    float*    agg   = (float*)alloc((size_t)N * 64 * 4);
    float*    gsum  = (float*)alloc((size_t)G * 64 * 4);
    int*      cnt   = (int*)alloc((size_t)G * 4);

    const int T = 256;
    const int bE  = (E + T - 1) / T;
    const int bN  = (N + T - 1) / T;
    const int bNH = (N * 64 + T - 1) / T;
    const int tilesN = (N + 15) / 16;
    const int tilesE = (E + 15) / 16;
    int gemmN = (tilesN + 7) / 8; if (gemmN > 512)  gemmN = 512;
    int gemmE = (tilesE + 7) / 8; if (gemmE > 1024) gemmE = 1024;

    // ---- init state (workspace is poisoned each run) ----
    k_fill_u32<<<bN, T, 0, stream>>>(m1, 0x7F800000u, N);          // +inf bits
    k_fill_u32<<<bN, T, 0, stream>>>(m2, 0x7F800000u, N);
    k_fill_u32<<<bN, T, 0, stream>>>(e1, 0xFFFFFFFFu, N);
    k_fill_u32<<<bN, T, 0, stream>>>(e2, 0xFFFFFFFFu, N);
    k_fill_u32<<<bN, T, 0, stream>>>((unsigned*)deg, 0u, N);
    k_fill_u32<<<bNH, T, 0, stream>>>((unsigned*)agg, 0u, N * 64);
    k_fill_u32<<<(G * 64 + T - 1) / T, T, 0, stream>>>((unsigned*)gsum, 0u, G * 64);
    k_fill_u32<<<1, T, 0, stream>>>((unsigned*)cnt, 0u, G);

    // ---- fragment-packed f16 weights ----
    k_pack_weight<<<16, T, 0, stream>>>(c1lw, wp_c1lw);
    k_pack_weight<<<16, T, 0, stream>>>(c1bw, wp_c1bw);
    k_pack_weight<<<16, T, 0, stream>>>(c2lw, wp_c2lw);
    k_pack_weight<<<16, T, 0, stream>>>(c2bw, wp_c2bw);
    k_pack_weight<<<16, T, 0, stream>>>(saw,  wp_saw);

    // ---- nearest / 2nd-nearest neighbor search ----
    k_edge_dist<<<bE, T, 0, stream>>>(pos, src, dst, dbits, m1, deg, E);
    k_e1<<<bE, T, 0, stream>>>(dbits, src, m1, e1, E);
    k_m2<<<bE, T, 0, stream>>>(dbits, src, e1, m2, E);
    k_e2<<<bE, T, 0, stream>>>(dbits, src, e1, m2, e2, E);
    k_nn<<<bN, T, 0, stream>>>(e1, e2, dst, nnA, nnB, N, E);

    // ---- per-edge geometry ----
    k_geo<<<bE, T, 0, stream>>>(pos, src, dst, nnA, nnB, deg, geo, E);

    // ---- conv layer 1 ----
    k_f32_to_f16<<<bNH, T, 0, stream>>>(x, hh, N * 64);
    node_gemm_f16<false><<<gemmN, T, 0, stream>>>(hh, wp_c1lw, c1lb, hlin, N);
    edge_conv_wmma<<<gemmE, T, 0, stream>>>(geo, c1aw, c1ab, wp_c1bw, c1bb,
                                            hlin, src, dst, agg, E);
    k_relu_f32_to_f16<<<bNH, T, 0, stream>>>(agg, hh, N * 64);

    // ---- conv layer 2 ----
    k_fill_u32<<<bNH, T, 0, stream>>>((unsigned*)agg, 0u, N * 64);
    node_gemm_f16<false><<<gemmN, T, 0, stream>>>(hh, wp_c2lw, c2lb, hlin, N);
    edge_conv_wmma<<<gemmE, T, 0, stream>>>(geo, c2aw, c2ab, wp_c2bw, c2bb,
                                            hlin, src, dst, agg, E);
    k_relu_f32_to_f16<<<bNH, T, 0, stream>>>(agg, hh, N * 64);

    // ---- saw layer (relu) into hlin ----
    node_gemm_f16<true><<<gemmN, T, 0, stream>>>(hh, wp_saw, sab, hlin, N);

    // ---- global mean pool + head ----
    k_pool_cnt<<<bN, T, 0, stream>>>(batch, cnt, N);
    k_pool_sum<<<bNH, T, 0, stream>>>(hlin, batch, gsum, N);
    k_final<<<1, 64, 0, stream>>>(gsum, cnt, l1w, l1b, l2w, l2b,
                                  (float*)d_out, G);
}
